// NemotronHMOE_2216203125390
// MI455X (gfx1250) — matrix operand order repaired and checked by reading, not compiled, not run on hardware
//
#include <hip/hip_runtime.h>
#include <cstdint>
#include <cstddef>

// ---------------- problem constants ----------------
#define H_DIM   1024
#define NEXP    64
#define KSEL    8
#define IMOE    512
#define ISH     2048
#define RSCALE  2.5f

// ---------------- tiling ----------------
#define BM      32                 // tokens per block (2 M-tiles of 16)
#define WT_LD   40                 // transposed B stage pitch (32 k + 8 pad) halfwords
#define LDX     (H_DIM + 8)        // X stage pitch (halfwords)   = 1032
#define LDY     (IMOE + 8)         // Y stage pitch (halfwords)   = 520

typedef __attribute__((ext_vector_type(16))) __bf16 v16bf;
typedef __attribute__((ext_vector_type(8)))  float  v8f;

union BFrag {
  v16bf v;
  unsigned short u[16];
  uint4 q[2];
};

__device__ __forceinline__ unsigned short f2bf(float f) {
  union { float f; unsigned u; } x; x.f = f;
  unsigned r = x.u + 0x7FFFu + ((x.u >> 16) & 1u);   // round-to-nearest-even
  return (unsigned short)(r >> 16);
}

__device__ __forceinline__ unsigned pk2bf(float a, float b) {
  return (unsigned)f2bf(a) | ((unsigned)f2bf(b) << 16);
}

// Stage rows [k0,k0+32) x cols [0,512) of W (row pitch ldw f32) transposed
// into lWt[col*WT_LD + k].  256 threads: each owns a 4(k) x 16(col) sub-tile,
// packs 4 k-values per column into one b64 LDS store.
__device__ __forceinline__ void stageB(const float* __restrict__ W, int ldw,
                                       int k0, int kmax, unsigned tid,
                                       unsigned short* lWt) {
  const int r4 = (int)(tid >> 5) * 4;        // k-row group: 0,4,...,28
  const int cb = (int)(tid & 31) * 16;       // col group: 0,16,...,496
  const float* src = W + (size_t)(k0 + r4) * (size_t)ldw + cb;
  if (k0 + 32 < kmax) __builtin_prefetch(src + (size_t)32 * ldw, 0, 1);
#pragma unroll
  for (int j = 0; j < 4; ++j) {
    float4 r0 = *(const float4*)(src + 0 * (size_t)ldw + j * 4);
    float4 r1 = *(const float4*)(src + 1 * (size_t)ldw + j * 4);
    float4 r2 = *(const float4*)(src + 2 * (size_t)ldw + j * 4);
    float4 r3 = *(const float4*)(src + 3 * (size_t)ldw + j * 4);
    unsigned short* dst = lWt + (size_t)(cb + j * 4) * WT_LD + r4;
    uint2 u;
    u.x = pk2bf(r0.x, r1.x); u.y = pk2bf(r2.x, r3.x);
    *(uint2*)(dst + 0 * WT_LD) = u;
    u.x = pk2bf(r0.y, r1.y); u.y = pk2bf(r2.y, r3.y);
    *(uint2*)(dst + 1 * WT_LD) = u;
    u.x = pk2bf(r0.z, r1.z); u.y = pk2bf(r2.z, r3.z);
    *(uint2*)(dst + 2 * WT_LD) = u;
    u.x = pk2bf(r0.w, r1.w); u.y = pk2bf(r2.w, r3.w);
    *(uint2*)(dst + 3 * WT_LD) = u;
  }
}

// Stage 32 gathered token rows of X (f32 global) into LDS as bf16, pitch LDX.
// 256 threads: thread owns rows (tid>>4) and (tid>>4)+16, 64-col segment.
__device__ __forceinline__ void stageX(const int* lTok,
                                       const float* __restrict__ x,
                                       unsigned tid, unsigned short* lX) {
#pragma unroll
  for (int rp = 0; rp < 2; ++rp) {
    const int row = (int)(tid >> 4) + rp * 16;
    const int seg = (int)(tid & 15) * 64;
    const float* src = x + (size_t)lTok[row] * H_DIM + seg;
    unsigned short* dst = lX + (size_t)row * LDX + seg;
#pragma unroll
    for (int j = 0; j < 64; j += 8) {
      float4 a = *(const float4*)(src + j);
      float4 b = *(const float4*)(src + j + 4);
      uint4 u;
      u.x = pk2bf(a.x, a.y); u.y = pk2bf(a.z, a.w);
      u.z = pk2bf(b.x, b.y); u.w = pk2bf(b.z, b.w);
      *(uint4*)(dst + j) = u;
    }
  }
}

// B fragment (32x16 bf16): lane holds column n=lane&15, u[0..7]=K half*8.., u[8..15]=K 16+half*8..
__device__ __forceinline__ v16bf loadBfrag(const unsigned short* lWt, int colBase,
                                           int lane) {
  int n = lane & 15, half = lane >> 4;
  const unsigned short* p = lWt + (size_t)(colBase + n) * WT_LD + half * 8;
  BFrag b;
  b.q[0] = *(const uint4*)(p);
  b.q[1] = *(const uint4*)(p + 16);
  return b.v;
}

// A fragment (16x32 bf16) from LDS, row pitch `pitch` halfwords:
// lane row m=lane&15, u[0..7]=K k0+half*8.. , u[8..15]=K k0+16+half*8..
__device__ __forceinline__ v16bf loadAfrag(const unsigned short* base, int pitch,
                                           int k0, int lane) {
  int m = lane & 15, half = lane >> 4;
  const unsigned short* p = base + (size_t)m * pitch + k0 + half * 8;
  BFrag a;
  a.q[0] = *(const uint4*)(p);
  a.q[1] = *(const uint4*)(p + 16);
  return a.v;
}

// ---------------- kernel 1: zero output + counters ----------------
__global__ void moe_zero_kernel(float* __restrict__ out, int n,
                                int* __restrict__ ints, int nint) {
  int i = (blockIdx.x * blockDim.x + threadIdx.x) * 4;
  if (i < n) {
    float4 z = {0.f, 0.f, 0.f, 0.f};
    *(float4*)(out + i) = z;
  }
  if (blockIdx.x == 0 && (int)threadIdx.x < nint) ints[threadIdx.x] = 0;
}

// ---------------- kernel 2: router ----------------
__global__ __launch_bounds__(64) void moe_router_kernel(
    const float* __restrict__ x, const float* __restrict__ rw,
    const float* __restrict__ bias, int* __restrict__ g_tidx,
    float* __restrict__ g_tw, int* __restrict__ counts) {
  __shared__ float sx[H_DIM];
  __shared__ float ssc[NEXP];   // sigmoid scores
  __shared__ float ssf[NEXP];   // score + bias
  __shared__ float sme[NEXP];   // masked
  __shared__ float sg[8];
  __shared__ int   sgm;

  int t = blockIdx.x;
  int tid = threadIdx.x;
  const float* xp = x + (size_t)t * H_DIM;
  for (int i = tid * 4; i < H_DIM; i += 64 * 4)
    *(float4*)&sx[i] = *(const float4*)&xp[i];
  __syncthreads();

  float acc = 0.f;
  const float* wr = rw + (size_t)tid * H_DIM;
  for (int h = 0; h < H_DIM; h += 4) {
    float4 w = *(const float4*)&wr[h];
    acc += w.x * sx[h] + w.y * sx[h + 1] + w.z * sx[h + 2] + w.w * sx[h + 3];
  }
  float sc = 1.f / (1.f + __expf(-acc));
  ssc[tid] = sc;
  ssf[tid] = sc + bias[tid];
  __syncthreads();

  if (tid < 8) {                      // group top-2 sums
    float m1 = -1e30f, m2 = -1e30f;
    for (int j = 0; j < 8; ++j) {
      float v = ssf[tid * 8 + j];
      if (v > m1) { m2 = m1; m1 = v; } else if (v > m2) m2 = v;
    }
    sg[tid] = m1 + m2;
  }
  __syncthreads();
  if (tid == 0) {                     // top-4 groups
    int mask = 0;
    for (int it = 0; it < 4; ++it) {
      float best = -1e30f; int bg = 0;
      for (int g = 0; g < 8; ++g)
        if (!((mask >> g) & 1) && sg[g] > best) { best = sg[g]; bg = g; }
      mask |= 1 << bg;
    }
    sgm = mask;
  }
  __syncthreads();
  sme[tid] = ((sgm >> (tid >> 3)) & 1) ? ssf[tid] : -1e30f;
  __syncthreads();
  if (tid == 0) {                     // top-8 experts + normalize
    int sel[KSEL]; float w[KSEL]; float sum = 0.f;
    for (int k = 0; k < KSEL; ++k) {
      float best = -1e30f; int be = 0;
      for (int e = 0; e < NEXP; ++e)
        if (sme[e] > best) { best = sme[e]; be = e; }
      sme[be] = -2e30f;
      sel[k] = be; w[k] = ssc[be]; sum += w[k];
    }
    float inv = RSCALE / (sum + 1e-20f);
    for (int k = 0; k < KSEL; ++k) {
      g_tidx[t * KSEL + k] = sel[k];
      g_tw[t * KSEL + k]   = w[k] * inv;
      atomicAdd(&counts[sel[k]], 1);
    }
  }
}

// ---------------- kernel 3: exclusive scan over 64 counts ----------------
__global__ void moe_scan_kernel(const int* __restrict__ counts,
                                int* __restrict__ offsets) {
  if (threadIdx.x == 0 && blockIdx.x == 0) {
    int o = 0;
    for (int e = 0; e < NEXP; ++e) { offsets[e] = o; o += counts[e]; }
  }
}

// ---------------- kernel 4: scatter token lists ----------------
__global__ void moe_scatter_kernel(const int* __restrict__ g_tidx,
                                   const float* __restrict__ g_tw,
                                   const int* __restrict__ offsets,
                                   int* __restrict__ cursor,
                                   int* __restrict__ toklist,
                                   float* __restrict__ tokw, int T) {
  int t = blockIdx.x * blockDim.x + threadIdx.x;
  if (t >= T) return;
  for (int k = 0; k < KSEL; ++k) {
    int e = g_tidx[t * KSEL + k];
    int p = atomicAdd(&cursor[e], 1);
    int pos = offsets[e] + p;
    toklist[pos] = t;
    tokw[pos]    = g_tw[t * KSEL + k];
  }
}

// ---------------- kernel 5: expert FFN (WMMA bf16) ----------------
// grid.y in [0,64): routed expert e ; [64,68): shared-expert I-chunk (512 wide)
// grid.x: token block of 32 (fast axis -> same-expert blocks co-resident in L2).
// 256 threads = 8 waves; wave owns 64 output cols x 32 rows (2 M x 4 N tiles).
// LDS: lWt 40 KB weight stage + lXY 66 KB (X bf16 in ph.1 / Y bf16 in ph.2).
__global__ __launch_bounds__(256) void moe_ffn_kernel(
    const float* __restrict__ x, const float* __restrict__ up_w,
    const float* __restrict__ down_w, const float* __restrict__ sh_up,
    const float* __restrict__ sh_down, const int* __restrict__ counts,
    const int* __restrict__ offsets, const int* __restrict__ toklist,
    const float* __restrict__ tokw, float* __restrict__ out) {
  __shared__ unsigned short lWt[512 * WT_LD];  // 40960 B transposed B stage
  __shared__ unsigned short lXY[BM * LDX];     // 66048 B: X (ph.1) / Y (ph.2)
  __shared__ int   lTok[BM];
  __shared__ float lWgt[BM];

  const int tb = blockIdx.x;                   // token block (fast axis)
  const int ex = blockIdx.y;                   // expert / shared chunk (slow axis)
  const unsigned tid = threadIdx.x;
  const int wave = (int)(tid >> 5), lane = (int)(tid & 31);
  const int m = lane & 15, half = lane >> 4;
  const int colW = wave * 64;

  const float* upB; int upLd; const float* dnB;
  const bool is_shared = (ex >= NEXP);
  int segBase = 0, rem = BM;
  if (!is_shared) {
    int cnt = counts[ex];
    if (tb * BM >= cnt) return;               // uniform early-exit, EXEC stays full
    rem = cnt - tb * BM;
    segBase = offsets[ex] + tb * BM;
    upB = up_w + (size_t)ex * H_DIM * IMOE;  upLd = IMOE;
    dnB = down_w + (size_t)ex * IMOE * H_DIM;
  } else {
    int c = ex - NEXP;                        // shared chunk 0..3
    upB = sh_up + c * 512;                   upLd = ISH;
    dnB = sh_down + (size_t)(c * 512) * H_DIM;
  }

  if ((int)tid < BM) {
    int tok; float w;
    if (is_shared) { tok = tb * BM + (int)tid; w = 1.f; }
    else if ((int)tid < rem) { tok = toklist[segBase + tid]; w = tokw[segBase + tid]; }
    else { tok = 0; w = 0.f; }                // padded rows contribute 0
    lTok[tid] = tok; lWgt[tid] = w;
  }
  __syncthreads();
  stageX(lTok, x, tid, lXY);                  // X -> bf16 LDS (once per block)

  // ---- Phase 1: Y[32,512] = relu2( X[32,1024] @ Wup[1024,512] ) ----
  v8f acc[2][4] = {};
  for (int k0 = 0; k0 < H_DIM; k0 += 32) {
    stageB(upB, upLd, k0, H_DIM, tid, lWt);
    __syncthreads();                          // covers stageX on first iter
    v16bf a0 = loadAfrag(lXY, LDX, k0, lane);
    v16bf a1 = loadAfrag(lXY + 16 * LDX, LDX, k0, lane);
    v16bf b0 = loadBfrag(lWt, colW +  0, lane);
    v16bf b1 = loadBfrag(lWt, colW + 16, lane);
    v16bf b2 = loadBfrag(lWt, colW + 32, lane);
    v16bf b3 = loadBfrag(lWt, colW + 48, lane);
    acc[0][0] = __builtin_amdgcn_wmma_f32_16x16x32_bf16(false, a0, false, b0, (short)0, acc[0][0], false, false);
    acc[0][1] = __builtin_amdgcn_wmma_f32_16x16x32_bf16(false, a0, false, b1, (short)0, acc[0][1], false, false);
    acc[0][2] = __builtin_amdgcn_wmma_f32_16x16x32_bf16(false, a0, false, b2, (short)0, acc[0][2], false, false);
    acc[0][3] = __builtin_amdgcn_wmma_f32_16x16x32_bf16(false, a0, false, b3, (short)0, acc[0][3], false, false);
    acc[1][0] = __builtin_amdgcn_wmma_f32_16x16x32_bf16(false, a1, false, b0, (short)0, acc[1][0], false, false);
    acc[1][1] = __builtin_amdgcn_wmma_f32_16x16x32_bf16(false, a1, false, b1, (short)0, acc[1][1], false, false);
    acc[1][2] = __builtin_amdgcn_wmma_f32_16x16x32_bf16(false, a1, false, b2, (short)0, acc[1][2], false, false);
    acc[1][3] = __builtin_amdgcn_wmma_f32_16x16x32_bf16(false, a1, false, b3, (short)0, acc[1][3], false, false);
    __syncthreads();
  }

  __syncthreads();                            // all X reads done before Y overwrite
#pragma unroll
  for (int mt = 0; mt < 2; ++mt)
#pragma unroll
    for (int nt = 0; nt < 4; ++nt)
#pragma unroll
      for (int i = 0; i < 8; ++i) {
        float v = acc[mt][nt][i];
        v = v > 0.f ? v * v : 0.f;            // relu^2
        int row = mt * 16 + i + 8 * half;
        lXY[(size_t)row * LDY + colW + nt * 16 + m] = f2bf(v);
      }

  // ---- Phase 2: out += w_t * ( Y[32,512] @ Wdown[512,1024] ), 2 col-panels ----
  for (int panel = 0; panel < 2; ++panel) {
    v8f acc2[2][4] = {};
    for (int k0 = 0; k0 < IMOE; k0 += 32) {
      __syncthreads();                        // guards lY writes & lWt reuse
      stageB(dnB + panel * 512, H_DIM, k0, IMOE, tid, lWt);
      __syncthreads();
      v16bf a0 = loadAfrag(lXY, LDY, k0, lane);
      v16bf a1 = loadAfrag(lXY + 16 * LDY, LDY, k0, lane);
      v16bf b0 = loadBfrag(lWt, colW +  0, lane);
      v16bf b1 = loadBfrag(lWt, colW + 16, lane);
      v16bf b2 = loadBfrag(lWt, colW + 32, lane);
      v16bf b3 = loadBfrag(lWt, colW + 48, lane);
      acc2[0][0] = __builtin_amdgcn_wmma_f32_16x16x32_bf16(false, a0, false, b0, (short)0, acc2[0][0], false, false);
      acc2[0][1] = __builtin_amdgcn_wmma_f32_16x16x32_bf16(false, a0, false, b1, (short)0, acc2[0][1], false, false);
      acc2[0][2] = __builtin_amdgcn_wmma_f32_16x16x32_bf16(false, a0, false, b2, (short)0, acc2[0][2], false, false);
      acc2[0][3] = __builtin_amdgcn_wmma_f32_16x16x32_bf16(false, a0, false, b3, (short)0, acc2[0][3], false, false);
      acc2[1][0] = __builtin_amdgcn_wmma_f32_16x16x32_bf16(false, a1, false, b0, (short)0, acc2[1][0], false, false);
      acc2[1][1] = __builtin_amdgcn_wmma_f32_16x16x32_bf16(false, a1, false, b1, (short)0, acc2[1][1], false, false);
      acc2[1][2] = __builtin_amdgcn_wmma_f32_16x16x32_bf16(false, a1, false, b2, (short)0, acc2[1][2], false, false);
      acc2[1][3] = __builtin_amdgcn_wmma_f32_16x16x32_bf16(false, a1, false, b3, (short)0, acc2[1][3], false, false);
    }
#pragma unroll
    for (int mt = 0; mt < 2; ++mt)
#pragma unroll
      for (int nt = 0; nt < 4; ++nt)
#pragma unroll
        for (int i = 0; i < 8; ++i) {
          int row = mt * 16 + i + 8 * half;
          float v = acc2[mt][nt][i] * lWgt[row];
          int col = panel * 512 + colW + nt * 16 + m;
          atomicAdd(&out[(size_t)lTok[row] * H_DIM + col], v);
        }
  }
}

// ---------------- host launcher ----------------
extern "C" void kernel_launch(void* const* d_in, const int* in_sizes, int n_in,
                              void* d_out, int out_size, void* d_ws, size_t ws_size,
                              hipStream_t stream) {
  const float* x   = (const float*)d_in[0];
  const float* rw  = (const float*)d_in[1];
  const float* cb  = (const float*)d_in[2];
  const float* up  = (const float*)d_in[3];
  const float* dn  = (const float*)d_in[4];
  const float* sup = (const float*)d_in[5];
  const float* sdn = (const float*)d_in[6];
  float* out = (float*)d_out;

  const int T = in_sizes[0] / H_DIM;          // 2048 tokens

  char* ws = (char*)d_ws;
  size_t o = 0;
  int*   tidx    = (int*)(ws + o);  o += (size_t)T * KSEL * sizeof(int);
  float* tw      = (float*)(ws + o); o += (size_t)T * KSEL * sizeof(float);
  int*   counts  = (int*)(ws + o);  o += 256;              // 64 used
  int*   cursor  = (int*)(ws + o);  o += 256;              // 64 used
  int*   offsets = (int*)(ws + o);  o += 256;              // 64 used
  int*   toklist = (int*)(ws + o);  o += (size_t)T * KSEL * sizeof(int);
  float* tokw    = (float*)(ws + o); o += (size_t)T * KSEL * sizeof(float);
  (void)o; (void)ws_size; (void)n_in; (void)out_size;

  const int nOut = T * H_DIM;
  moe_zero_kernel<<<dim3((nOut / 4 + 255) / 256), dim3(256), 0, stream>>>(
      out, nOut, counts, 128);                // zeroes counts+cursor (contiguous)
  moe_router_kernel<<<dim3(T), dim3(64), 0, stream>>>(x, rw, cb, tidx, tw, counts);
  moe_scan_kernel<<<dim3(1), dim3(1), 0, stream>>>(counts, offsets);
  moe_scatter_kernel<<<dim3((T + 255) / 256), dim3(256), 0, stream>>>(
      tidx, tw, offsets, cursor, toklist, tokw, T);
  moe_ffn_kernel<<<dim3(T / BM, NEXP + 4), dim3(256), 0, stream>>>(
      x, up, dn, sup, sdn, counts, offsets, toklist, tokw, out);
}
